// Head_24988119728412
// MI455X (gfx1250) — matrix operand order repaired and checked by reading, not compiled, not run on hardware
//
#include <hip/hip_runtime.h>
#include <hip/hip_bf16.h>

constexpr int Bn  = 8;
constexpr int Tn  = 2048;
constexpr int Cn  = 1024;
constexpr int HSn = 64;
constexpr int BTn = Bn * Tn;

typedef __attribute__((ext_vector_type(16))) __bf16    v16bf;
typedef __attribute__((ext_vector_type(8)))  float     v8f;
typedef __attribute__((ext_vector_type(4)))  unsigned  u32x4;
typedef __attribute__((ext_vector_type(8)))  int       i32x8;
typedef __attribute__((ext_vector_type(4)))  int       i32x4;

union Frag16 {
  v16bf    v;
  int4     q[2];
  unsigned u[8];
};

__device__ inline v8f wmma_bf16(v16bf a, v16bf b, v8f c) {
  return __builtin_amdgcn_wmma_f32_16x16x32_bf16(false, a, false, b,
                                                 (short)0, c, false, false);
}

__device__ inline unsigned pk2(float a, float b) {
  union { __bf16 h[2]; unsigned u; } x;
  x.h[0] = (__bf16)a; x.h[1] = (__bf16)b;
  return x.u;
}

__device__ inline v8f vzero() {
  v8f z = {0.f, 0.f, 0.f, 0.f, 0.f, 0.f, 0.f, 0.f};
  return z;
}

// ---------------------------------------------------------------------------
// Prep: transpose + convert the three weight matrices to bf16 W^T [3][64][1024]
// ---------------------------------------------------------------------------
__global__ __launch_bounds__(256) void wtrans_kernel(
    const float* __restrict__ W0, const float* __restrict__ W1,
    const float* __restrict__ W2, __bf16* __restrict__ wT) {
  int idx = blockIdx.x * 256 + threadIdx.x;  // 3*1024*64 elements
  if (idx >= 3 * Cn * HSn) return;
  int w = idx / (Cn * HSn);
  int r = idx % (Cn * HSn);
  int c = r / HSn;
  int h = r % HSn;
  const float* W = (w == 0) ? W0 : (w == 1) ? W1 : W2;
  wT[(size_t)w * HSn * Cn + (size_t)h * Cn + c] = (__bf16)W[r];
}

// ---------------------------------------------------------------------------
// Projection: R[h][t] = sum_c W[c][h] * X[t][c]  (== proj[t][h]), bf16 out.
// A = W^T staged into LDS via the Tensor Data Mover (bf16, pre-transposed in
// global), B = X^T read directly from the row-major f32 input.
// dstRM != null : store row-major [t][h]   (q, k)
// dstRM == null : store transposed [h][t] via LDS bounce (v)
// ---------------------------------------------------------------------------
#define KC    256
#define WPAD  (KC + 8)          // 264 elems = 528 B row stride (16B aligned)
#define VPAD  (128 + 8)

__global__ __launch_bounds__(256) void proj_kernel(
    const float* __restrict__ X, const __bf16* __restrict__ WT,
    __bf16* __restrict__ dstRM, __bf16* __restrict__ dstTM) {
  __shared__ __bf16 wT[HSn][WPAD];     // W^T chunk, bf16 (TDM destination)
  __shared__ __bf16 vstage[HSn][VPAD]; // transpose bounce for v

  const int tid  = threadIdx.x;
  const int lane = tid & 31;
  const int wave = tid >> 5;
  const int l16  = lane & 15;
  const int up   = lane >> 4;

  const int tblk = blockIdx.x * 128;
  const int tg   = tblk + wave * 16 + l16;   // global row (N dim)
  const int tl   = wave * 16 + l16;

  v8f acc[4] = {vzero(), vzero(), vzero(), vzero()};

  const unsigned lds_off = (unsigned)(size_t)(&wT[0][0]); // LDS byte offset

  for (int cc = 0; cc < Cn; cc += KC) {
    __syncthreads();   // previous chunk fully consumed
    if (wave == 0) {
      // TDM: load W^T tile [64 rows][256 cols] (bf16) into LDS with padding:
      // pad_interval=6 (every 128 dwords = one 512B row), pad_amount=3 (4
      // dwords = 16B) -> LDS row stride 528B, matching wT[64][264].
      unsigned long long ga =
          (unsigned long long)(size_t)(WT + cc);   // tile start (bytes)
      u32x4 g0 = { 1u,                             // count=1, no gather
                   lds_off,                        // lds_addr
                   (unsigned)(ga & 0xFFFFFFFFu),
                   (unsigned)((ga >> 32) & 0x1FFFFFFu) | 0x80000000u }; // type=2
      i32x8 g1 = { (1 << 16) | (1 << 20) | (6 << 22) | (3 << 25),
                   // ^ data_size=2B, pad_enable, pad_interval=6, pad_amount=3
                   (int)(Cn << 16),   // tensor_dim0 = 1024 (bits 79:48)
                   (int)(HSn << 16),  // tensor_dim1 = 64   (bits 111:80)
                   (int)(KC << 16),   // tile_dim0 = 256    (bits 127:112)
                   HSn,               // tile_dim1 = 64     (bits 143:128)
                   Cn,                // tensor_dim0_stride = 1024 (207:160)
                   0, 0 };
      i32x4 gz4 = { 0, 0, 0, 0 };
      i32x8 gz8 = { 0, 0, 0, 0, 0, 0, 0, 0 };
      __builtin_amdgcn_tensor_load_to_lds(g0, g1, gz4, gz4, gz8, 0);
      __builtin_amdgcn_s_wait_tensorcnt(0);
    }
    __syncthreads();   // chunk visible to all waves

    for (int kk = 0; kk < KC; kk += 32) {
      // B frag (X^T 32c x 16t): lane = column t, contiguous run of 16 f32
      const float* xr = X + (size_t)tg * Cn + (cc + kk + up * 16);
      float4 x0 = *(const float4*)(xr + 0);
      float4 x1 = *(const float4*)(xr + 4);
      float4 x2 = *(const float4*)(xr + 8);
      float4 x3 = *(const float4*)(xr + 12);
      Frag16 bx;
      bx.v[0]  = (__bf16)x0.x; bx.v[1]  = (__bf16)x0.y;
      bx.v[2]  = (__bf16)x0.z; bx.v[3]  = (__bf16)x0.w;
      bx.v[4]  = (__bf16)x1.x; bx.v[5]  = (__bf16)x1.y;
      bx.v[6]  = (__bf16)x1.z; bx.v[7]  = (__bf16)x1.w;
      bx.v[8]  = (__bf16)x2.x; bx.v[9]  = (__bf16)x2.y;
      bx.v[10] = (__bf16)x2.z; bx.v[11] = (__bf16)x2.w;
      bx.v[12] = (__bf16)x3.x; bx.v[13] = (__bf16)x3.y;
      bx.v[14] = (__bf16)x3.z; bx.v[15] = (__bf16)x3.w;

      // load all four A-frags first so the DS loads can overlap the WMMAs
      const int klo = kk + (up ? 16 : 0);
      Frag16 aw[4];
#pragma unroll
      for (int ht = 0; ht < 4; ht++) {
        int h = ht * 16 + l16;
        aw[ht].q[0] = *(const int4*)&wT[h][klo];
        aw[ht].q[1] = *(const int4*)&wT[h][klo + 8];
      }
#pragma unroll
      for (int ht = 0; ht < 4; ht++)
        acc[ht] = wmma_bf16(aw[ht].v, bx.v, acc[ht]);
    }
  }

  if (dstRM) {
    // D layout: lane holds fixed t, h contiguous -> one b128 per frag
#pragma unroll
    for (int ht = 0; ht < 4; ht++) {
      int4 p;
      p.x = (int)pk2(acc[ht][0], acc[ht][1]);
      p.y = (int)pk2(acc[ht][2], acc[ht][3]);
      p.z = (int)pk2(acc[ht][4], acc[ht][5]);
      p.w = (int)pk2(acc[ht][6], acc[ht][7]);
      *(int4*)(dstRM + (size_t)tg * HSn + ht * 16 + up * 8) = p;
    }
  } else {
    // bounce through LDS to store v transposed as [b][h][t]
#pragma unroll
    for (int ht = 0; ht < 4; ht++)
#pragma unroll
      for (int i = 0; i < 8; i++)
        vstage[ht * 16 + up * 8 + i][tl] = (__bf16)acc[ht][i];
    __syncthreads();
    int bb  = tblk >> 11;
    int tt0 = tblk & (Tn - 1);
    for (int j = tid; j < HSn * 16; j += 256) {
      int h  = j >> 4;
      int to = (j & 15) * 8;
      int4 val = *(const int4*)&vstage[h][to];
      *(int4*)(dstTM + ((size_t)bb * HSn + h) * Tn + tt0 + to) = val;
    }
  }
}

// ---------------------------------------------------------------------------
// Flash attention (transposed): per 16-query wave,
//   S' = K_tile · Q^T  (M=s, N=q)  -> lane owns a fixed query
//   online softmax (per-lane stats, one shfl_xor(16) per block)
//   O^T += V^T · P^T   (M=h, N=q), P^T rebuilt in registers via shuffles
// ---------------------------------------------------------------------------
__global__ __launch_bounds__(128) void attn_kernel(
    const __bf16* __restrict__ qm, const __bf16* __restrict__ km,
    const __bf16* __restrict__ vT, const int* __restrict__ mask,
    float* __restrict__ out) {
  const int lane = threadIdx.x & 31;
  const int wave = threadIdx.x >> 5;
  const int l16  = lane & 15;
  const int up   = lane >> 4;

  const int tblk = blockIdx.x * 64 + wave * 16;
  const int b    = tblk >> 11;
  const int tq0  = tblk & (Tn - 1);
  const int qg   = tblk + l16;
  const int qloc = tq0 + l16;

  const int*    maskb = mask + (size_t)b * Tn;
  const __bf16* kb    = km + (size_t)b * Tn * HSn;
  const __bf16* vb    = vT + (size_t)b * HSn * Tn;

  Frag16 bq[2];
  const __bf16* qrow = qm + (size_t)qg * HSn;
#pragma unroll
  for (int f = 0; f < 2; f++) {
    int hb = f * 32 + up * 8;
    bq[f].q[0] = *(const int4*)(qrow + hb);
    bq[f].q[1] = *(const int4*)(qrow + hb + 16);
  }

  float m_run = -1e30f, l_run = 0.0f;
  v8f o[4] = {vzero(), vzero(), vzero(), vzero()};

  const int smax = tq0 + 15;
  for (int s0 = 0; s0 <= smax; s0 += 32) {
    float pv[16];
    float mloc = -1e30f;
#pragma unroll
    for (int fi = 0; fi < 2; fi++) {
      const int sb = s0 + fi * 16;
      const __bf16* krow = kb + (size_t)(sb + l16) * HSn;
      v8f c = vzero();
#pragma unroll
      for (int hf = 0; hf < 2; hf++) {
        Frag16 ak;
        int klo = hf * 32 + up * 8;
        ak.q[0] = *(const int4*)(krow + klo);
        ak.q[1] = *(const int4*)(krow + klo + 16);
        c = wmma_bf16(ak.v, bq[hf].v, c);
      }
      const int se = sb + up * 8;
      int4 mv0 = *(const int4*)(maskb + se);
      int4 mv1 = *(const int4*)(maskb + se + 4);
      int mvals[8] = {mv0.x, mv0.y, mv0.z, mv0.w, mv1.x, mv1.y, mv1.z, mv1.w};
#pragma unroll
      for (int i = 0; i < 8; i++) {
        int s = se + i;
        float val = c[i] * 0.125f;
        bool ok = (s <= qloc) && (mvals[i] != 0);
        val = ok ? val : -1e30f;
        pv[fi * 8 + i] = val;
        mloc = fmaxf(mloc, val);
      }
    }
    mloc = fmaxf(mloc, __shfl_xor(mloc, 16, 32));
    float m_new = fmaxf(m_run, mloc);
    float alpha = __expf(m_run - m_new);
    float ssum = 0.0f;
#pragma unroll
    for (int j = 0; j < 16; j++) { pv[j] = __expf(pv[j] - m_new); ssum += pv[j]; }
    ssum += __shfl_xor(ssum, 16, 32);
    l_run = l_run * alpha + ssum;
    m_run = m_new;
#pragma unroll
    for (int ht = 0; ht < 4; ht++) o[ht] *= alpha;

    unsigned d0[4], d1[4];
#pragma unroll
    for (int j2 = 0; j2 < 4; j2++) {
      d0[j2] = pk2(pv[2 * j2],     pv[2 * j2 + 1]);
      d1[j2] = pk2(pv[8 + 2 * j2], pv[8 + 2 * j2 + 1]);
    }
    Frag16 bp;
#pragma unroll
    for (int j2 = 0; j2 < 4; j2++) {
      unsigned o0 = (unsigned)__shfl_xor((int)d0[j2], 16, 32);
      unsigned o1 = (unsigned)__shfl_xor((int)d1[j2], 16, 32);
      bp.u[j2]     = up ? o1 : d0[j2];
      bp.u[4 + j2] = up ? d1[j2] : o0;
    }

#pragma unroll
    for (int ht = 0; ht < 4; ht++) {
      int h = ht * 16 + l16;
      const __bf16* vrow = vb + (size_t)h * Tn + s0 + up * 16;
      Frag16 av;
      av.q[0] = *(const int4*)(vrow);
      av.q[1] = *(const int4*)(vrow + 8);
      o[ht] = wmma_bf16(av.v, bp.v, o[ht]);
    }
  }

  float inv = 1.0f / l_run;
  float* orow = out + (size_t)qg * HSn;
#pragma unroll
  for (int ht = 0; ht < 4; ht++) {
    int hb = ht * 16 + up * 8;
    float4 lo = make_float4(o[ht][0] * inv, o[ht][1] * inv,
                            o[ht][2] * inv, o[ht][3] * inv);
    float4 hi = make_float4(o[ht][4] * inv, o[ht][5] * inv,
                            o[ht][6] * inv, o[ht][7] * inv);
    *(float4*)(orow + hb)     = lo;
    *(float4*)(orow + hb + 4) = hi;
  }
}

extern "C" void kernel_launch(void* const* d_in, const int* in_sizes, int n_in,
                              void* d_out, int out_size, void* d_ws, size_t ws_size,
                              hipStream_t stream) {
  (void)in_sizes; (void)n_in; (void)out_size; (void)ws_size;
  const float* q_vec = (const float*)d_in[0];
  const float* k_vec = (const float*)d_in[1];
  const float* v_vec = (const float*)d_in[2];
  const int*   mask  = (const int*)d_in[3];
  const float* Wq    = (const float*)d_in[4];
  const float* Wk    = (const float*)d_in[5];
  const float* Wv    = (const float*)d_in[6];
  float* out = (float*)d_out;

  // workspace: q (2 MB) | k (2 MB) | v^T (2 MB) | W^T bf16 [3][64][1024]
  __bf16* qbf = (__bf16*)d_ws;
  __bf16* kbf = qbf + (size_t)BTn * HSn;
  __bf16* vTb = kbf + (size_t)BTn * HSn;
  __bf16* wTs = vTb + (size_t)BTn * HSn;

  wtrans_kernel<<<dim3((3 * Cn * HSn + 255) / 256), 256, 0, stream>>>(
      Wq, Wk, Wv, wTs);

  dim3 pgrid(BTn / 128);
  proj_kernel<<<pgrid, 256, 0, stream>>>(q_vec, wTs + 0 * Cn * HSn, qbf, nullptr);
  proj_kernel<<<pgrid, 256, 0, stream>>>(k_vec, wTs + 1 * Cn * HSn, kbf, nullptr);
  proj_kernel<<<pgrid, 256, 0, stream>>>(v_vec, wTs + 2 * Cn * HSn, nullptr, vTb);

  attn_kernel<<<dim3(BTn / 64), 128, 0, stream>>>(qbf, kbf, vTb, mask, out);
}
